// InductiveLayer_14388140442300
// MI455X (gfx1250) — matrix operand order repaired and checked by reading, not compiled, not run on hardware
//
#include <hip/hip_runtime.h>

// ---------------------------------------------------------------------------
// Problem constants (match reference)
// ---------------------------------------------------------------------------
static constexpr int Nn   = 50000;   // nodes
static constexpr int Fdim = 512;     // input features
static constexpr int Odim = 256;     // output dim
static constexpr int Hh   = 4;       // hops
static constexpr int Ee   = 800000;  // edges per hop
static constexpr int TwoO = 512;     // 2*O

// ---------------------------------------------------------------------------
// Vector types for WMMA (wave32, gfx1250)
// ---------------------------------------------------------------------------
typedef __attribute__((ext_vector_type(16))) __bf16 bf16x16;
typedef __attribute__((ext_vector_type(8)))  __bf16 bf16x8;
typedef __attribute__((ext_vector_type(8)))  float  f32x8;

union FragBF16 { bf16x16 v; bf16x8 h[2]; };

// ---------------------------------------------------------------------------
// f32 -> bf16 straight conversion (grid-stride)
// ---------------------------------------------------------------------------
__global__ __launch_bounds__(256)
void cvt_f32_bf16(const float* __restrict__ in, __bf16* __restrict__ out, size_t n) {
    size_t i = (size_t)blockIdx.x * blockDim.x + threadIdx.x;
    size_t stride = (size_t)gridDim.x * blockDim.x;
    for (; i < n; i += stride) out[i] = (__bf16)in[i];
}

// ---------------------------------------------------------------------------
// f32 [R,C] -> bf16 [C,R] transpose-convert; batched via blockIdx.z
// ---------------------------------------------------------------------------
__global__ __launch_bounds__(256)
void transpose_f32_bf16(const float* __restrict__ in, __bf16* __restrict__ out,
                        int R, int C) {
    size_t boff = (size_t)blockIdx.z * R * C;
    int idx = blockIdx.x * 256 + threadIdx.x;
    if (idx < R * C) {
        int r = idx / C, c = idx % C;
        out[boff + (size_t)c * R + r] = (__bf16)in[boff + idx];
    }
}

// ---------------------------------------------------------------------------
// Fold alpha/Wbase/Wr into a single transposed bf16 [O,O] matrix:
//   Wc = (sum_h clip(a_h)) * Wbase + sum_h clip(a_h) * Wr[h];  store Wc^T
// ---------------------------------------------------------------------------
__global__ __launch_bounds__(256)
void build_wct(const float* __restrict__ Wbase, const float* __restrict__ Wr,
               const float* __restrict__ alpha, __bf16* __restrict__ Wct) {
    int idx = blockIdx.x * 256 + threadIdx.x;
    if (idx >= Odim * Odim) return;
    int o = idx / Odim, p = idx % Odim;
    float s = 0.f, acc = 0.f;
#pragma unroll
    for (int h = 0; h < Hh; ++h) {
        float a = fminf(fmaxf(alpha[h], 0.f), 1.f);
        s += a;
        acc += a * Wr[(size_t)h * Odim * Odim + idx];
    }
    acc += s * Wbase[idx];
    Wct[(size_t)p * Odim + o] = (__bf16)acc;
}

// ---------------------------------------------------------------------------
// WMMA GEMM:  C[M,Nc] = A[M,K] * Bt[Nc,K]^T  (+bias, +relu), f32 accumulate.
// A, Bt bf16 row-major. 256 threads = 8 waves in 4(M) x 2(N) grid; each wave
// computes a 32x64 tile = 8 x v_wmma_f32_16x16x32_bf16 accumulators.
// Per K-step: 12x16B loads feed 8 WMMAs (24 load-bytes per WMMA).
// Operands are L2-resident (192MB L2; A slabs 51MB, weights <=512KB) so no
// LDS staging is required.
// ---------------------------------------------------------------------------
template <typename OutT, bool RELU, bool HAS_BIAS>
__global__ __launch_bounds__(256)
void wmma_gemm(const __bf16* __restrict__ A, const __bf16* __restrict__ Bt,
               const float* __restrict__ bias, OutT* __restrict__ C,
               int M, int K, int Nc) {
    const int lane = threadIdx.x & 31;
    const int wid  = threadIdx.x >> 5;     // 0..7
    const int wm   = wid & 3;              // M direction (0..3)
    const int wn   = wid >> 2;             // N direction (0..1)
    const int half = lane >> 4;
    const int l    = lane & 15;

    const int m0 = blockIdx.x * 128 + wm * 32;
    const int n0 = blockIdx.y * 128 + wn * 64;

    // A fragment rows (clamped for the ragged last M block; stores are guarded)
    int ra0 = m0 + l;      if (ra0 > M - 1) ra0 = M - 1;
    int ra1 = m0 + 16 + l; if (ra1 > M - 1) ra1 = M - 1;
    // A k order used by this code (same as B): lanes 0-15 -> K 0..15, lanes 16-31 -> K 16..31 (row = l)
    const __bf16* a0p = A + (size_t)ra0 * K + half * 16;
    const __bf16* a1p = A + (size_t)ra1 * K + half * 16;
    // B k order used by this code (same as A): lanes 0-15 -> K 0..15, lanes 16-31 -> K 16..31 (col = l)
    const __bf16* b0p = Bt + (size_t)(n0 + l)      * K + half * 16;
    const __bf16* b1p = Bt + (size_t)(n0 + 16 + l) * K + half * 16;
    const __bf16* b2p = Bt + (size_t)(n0 + 32 + l) * K + half * 16;
    const __bf16* b3p = Bt + (size_t)(n0 + 48 + l) * K + half * 16;

    f32x8 acc[2][4];
#pragma unroll
    for (int i = 0; i < 2; ++i)
#pragma unroll
        for (int j = 0; j < 4; ++j) acc[i][j] = (f32x8){};

    for (int k = 0; k < K; k += 32) {
        FragBF16 a0, a1, b0, b1, b2, b3;
        a0.h[0] = *(const bf16x8*)(a0p + k);
        a0.h[1] = *(const bf16x8*)(a0p + k + 8);
        a1.h[0] = *(const bf16x8*)(a1p + k);
        a1.h[1] = *(const bf16x8*)(a1p + k + 8);
        b0.h[0] = *(const bf16x8*)(b0p + k);
        b0.h[1] = *(const bf16x8*)(b0p + k + 8);
        b1.h[0] = *(const bf16x8*)(b1p + k);
        b1.h[1] = *(const bf16x8*)(b1p + k + 8);
        b2.h[0] = *(const bf16x8*)(b2p + k);
        b2.h[1] = *(const bf16x8*)(b2p + k + 8);
        b3.h[0] = *(const bf16x8*)(b3p + k);
        b3.h[1] = *(const bf16x8*)(b3p + k + 8);

        acc[0][0] = __builtin_amdgcn_wmma_f32_16x16x32_bf16(false, a0.v, false, b0.v,
                                                            (short)0, acc[0][0], false, false);
        acc[0][1] = __builtin_amdgcn_wmma_f32_16x16x32_bf16(false, a0.v, false, b1.v,
                                                            (short)0, acc[0][1], false, false);
        acc[0][2] = __builtin_amdgcn_wmma_f32_16x16x32_bf16(false, a0.v, false, b2.v,
                                                            (short)0, acc[0][2], false, false);
        acc[0][3] = __builtin_amdgcn_wmma_f32_16x16x32_bf16(false, a0.v, false, b3.v,
                                                            (short)0, acc[0][3], false, false);
        acc[1][0] = __builtin_amdgcn_wmma_f32_16x16x32_bf16(false, a1.v, false, b0.v,
                                                            (short)0, acc[1][0], false, false);
        acc[1][1] = __builtin_amdgcn_wmma_f32_16x16x32_bf16(false, a1.v, false, b1.v,
                                                            (short)0, acc[1][1], false, false);
        acc[1][2] = __builtin_amdgcn_wmma_f32_16x16x32_bf16(false, a1.v, false, b2.v,
                                                            (short)0, acc[1][2], false, false);
        acc[1][3] = __builtin_amdgcn_wmma_f32_16x16x32_bf16(false, a1.v, false, b3.v,
                                                            (short)0, acc[1][3], false, false);
    }

    // C/D 16x16 f32 layout: VGPR v, lane<16 -> (M=v,   N=l)
    //                              lane>=16 -> (M=v+8, N=l)
    auto store_tile = [&](const f32x8& a, int tm, int tn) {
        const int c = tn + l;
        const float b = HAS_BIAS ? bias[c] : 0.f;
        const int rbase = tm + half * 8;
#pragma unroll
        for (int v = 0; v < 8; ++v) {
            int r = rbase + v;
            if (r < M) {
                float x = a[v] + b;
                if (RELU) x = fmaxf(x, 0.f);
                C[(size_t)r * Nc + c] = (OutT)x;
            }
        }
    };
#pragma unroll
    for (int i = 0; i < 2; ++i)
#pragma unroll
        for (int j = 0; j < 4; ++j)
            store_tile(acc[i][j], m0 + i * 16, n0 + j * 16);
}

// ---------------------------------------------------------------------------
// COO scatter: one wave per edge, 8 f32 atomics per lane (256 cols).
// proj slab (51MB) is L2-resident; atomics resolve in L2 atomic units.
// ---------------------------------------------------------------------------
__global__ __launch_bounds__(256)
void edge_scatter(const int* __restrict__ rows, const int* __restrict__ cols,
                  const float* __restrict__ vals, const float* __restrict__ proj,
                  float* __restrict__ acc, int E) {
    int e = blockIdx.x * 8 + (int)(threadIdx.x >> 5);
    if (e >= E) return;
    int lane = threadIdx.x & 31;
    int r = rows[e], c = cols[e];
    float v = vals[e];
    const float4* src = (const float4*)(proj + (size_t)c * Odim + lane * 8);
    float* dst = acc + (size_t)r * Odim + lane * 8;
    float4 s0 = src[0], s1 = src[1];
    atomicAdd(dst + 0, v * s0.x);
    atomicAdd(dst + 1, v * s0.y);
    atomicAdd(dst + 2, v * s0.z);
    atomicAdd(dst + 3, v * s0.w);
    atomicAdd(dst + 4, v * s1.x);
    atomicAdd(dst + 5, v * s1.y);
    atomicAdd(dst + 6, v * s1.z);
    atomicAdd(dst + 7, v * s1.w);
}

__global__ __launch_bounds__(256)
void relu_inplace(float* __restrict__ x, size_t n) {
    size_t i = (size_t)blockIdx.x * blockDim.x + threadIdx.x;
    size_t stride = (size_t)gridDim.x * blockDim.x;
    for (; i < n; i += stride) x[i] = fmaxf(x[i], 0.f);
}

// ---------------------------------------------------------------------------
// Launcher
// ---------------------------------------------------------------------------
extern "C" void kernel_launch(void* const* d_in, const int* in_sizes, int n_in,
                              void* d_out, int out_size, void* d_ws, size_t ws_size,
                              hipStream_t stream) {
    const float* X     = (const float*)d_in[0];   // [N,F]
    const int*   rows  = (const int*)  d_in[1];   // [H,E]
    const int*   cols  = (const int*)  d_in[2];   // [H,E]
    const float* vals  = (const float*)d_in[3];   // [H,E]
    const float* W1    = (const float*)d_in[4];   // [F,2O]
    const float* b1    = (const float*)d_in[5];   // [2O]
    const float* W2    = (const float*)d_in[6];   // [2O,O]
    const float* b2    = (const float*)d_in[7];   // [O]
    const float* Wbase = (const float*)d_in[8];   // [O,O]
    const float* Wf    = (const float*)d_in[9];   // [H,F,O]
    const float* Wr    = (const float*)d_in[10];  // [H,O,O]
    const float* alpha = (const float*)d_in[11];  // [H]
    float* out = (float*)d_out;                   // [N,O] f32, doubles as accumulator

    // Carve workspace (256B aligned slabs)
    char* wsp = (char*)d_ws;
    auto carve = [&](size_t bytes) -> void* {
        void* p = (void*)wsp;
        wsp += (bytes + 255) & ~(size_t)255;
        return p;
    };
    __bf16* Xbf  = (__bf16*)carve((size_t)Nn * Fdim * 2);    // 51.2 MB
    __bf16* Hmid = (__bf16*)carve((size_t)Nn * TwoO * 2);    // 51.2 MB (reused as proj)
    __bf16* emb  = (__bf16*)carve((size_t)Nn * Odim * 2);    // 25.6 MB
    __bf16* W1t  = (__bf16*)carve((size_t)TwoO * Fdim * 2);  // [2O,F]
    __bf16* W2t  = (__bf16*)carve((size_t)Odim * TwoO * 2);  // [O,2O]
    __bf16* Wct  = (__bf16*)carve((size_t)Odim * Odim * 2);  // [O,O]
    __bf16* Wft  = (__bf16*)carve((size_t)Hh * Odim * Fdim * 2); // [H][O,F]
    float*  proj = (float*)Hmid;  // alias: Hmid dead after GEMM2; proj f32 slab same size

    const int mblk = (Nn + 127) / 128;  // 391

    // --- one-time precision conversions ---------------------------------
    cvt_f32_bf16<<<4096, 256, 0, stream>>>(X, Xbf, (size_t)Nn * Fdim);
    transpose_f32_bf16<<<dim3((Fdim * TwoO + 255) / 256, 1, 1), 256, 0, stream>>>(
        W1, W1t, Fdim, TwoO);
    transpose_f32_bf16<<<dim3((TwoO * Odim + 255) / 256, 1, 1), 256, 0, stream>>>(
        W2, W2t, TwoO, Odim);
    transpose_f32_bf16<<<dim3((Fdim * Odim + 255) / 256, 1, Hh), 256, 0, stream>>>(
        Wf, Wft, Fdim, Odim);
    build_wct<<<(Odim * Odim + 255) / 256, 256, 0, stream>>>(Wbase, Wr, alpha, Wct);

    // --- embedding MLP ---------------------------------------------------
    // Hmid = relu(X @ W1 + b1)          [N, 2O] bf16
    wmma_gemm<__bf16, true, true><<<dim3(mblk, TwoO / 128), 256, 0, stream>>>(
        Xbf, W1t, b1, Hmid, Nn, Fdim, TwoO);
    // emb = Hmid @ W2 + b2              [N, O] bf16
    wmma_gemm<__bf16, false, true><<<dim3(mblk, Odim / 128), 256, 0, stream>>>(
        Hmid, W2t, b2, emb, Nn, TwoO, Odim);
    // out = emb @ Wc   (folds base + all a_h * resid_h; fully writes d_out)
    wmma_gemm<float, false, false><<<dim3(mblk, Odim / 128), 256, 0, stream>>>(
        emb, Wct, nullptr, out, Nn, Odim, Odim);

    // --- per-hop: proj GEMM + COO scatter --------------------------------
    for (int h = 0; h < Hh; ++h) {
        wmma_gemm<float, false, false><<<dim3(mblk, Odim / 128), 256, 0, stream>>>(
            Xbf, Wft + (size_t)h * Odim * Fdim, nullptr, proj, Nn, Fdim, Odim);
        edge_scatter<<<(Ee + 7) / 8, 256, 0, stream>>>(
            rows + (size_t)h * Ee, cols + (size_t)h * Ee, vals + (size_t)h * Ee,
            proj, out, Ee);
    }

    // --- final relu ------------------------------------------------------
    relu_inplace<<<4096, 256, 0, stream>>>(out, (size_t)Nn * Odim);
}
